// RNNEncoder_60370060312695
// MI455X (gfx1250) — matrix operand order repaired and checked by reading, not compiled, not run on hardware
//
#include <hip/hip_runtime.h>
#include <hip/hip_bf16.h>
#include <math.h>

// ---------------------------------------------------------------------------
// RNN encoder on gfx1250: bf16 WMMA GEMMs + persistent recurrent scan
// ---------------------------------------------------------------------------

typedef __attribute__((ext_vector_type(16))) __bf16 v16bf;
typedef __attribute__((ext_vector_type(8)))  __bf16 v8bf;
typedef __attribute__((ext_vector_type(8)))  float  v8f;

#define E_  256
#define H_  1024
#define B_  32
#define S_  512
#define NWG 32                       // persistent workgroups in scan kernel
#define COLS_PER_WG (H_ / NWG)       // 32 columns of H per workgroup

// workspace layout (bytes)
#define WXT_OFF  0                           // WxT[n][k] bf16, [H][E]
#define WHT_OFF  (WXT_OFF + H_*E_*2)         // WhT[n][k] bf16, [H][H]
#define HB0_OFF  (WHT_OFF + (size_t)H_*H_*2) // h ping  bf16, [B][H]
#define HB1_OFF  (HB0_OFF + B_*H_*2)         // h pong  bf16, [B][H]
#define SYNC_OFF (HB1_OFF + B_*H_*2)         // u32 counter, u32 epoch

union V16 { v16bf v; v8bf h[2]; };

// ---------------------------------------------------------------------------
// Prep: bf16-convert + transpose weights, convert initial hidden, reset sync.
// ---------------------------------------------------------------------------
__global__ __launch_bounds__(256) void prep_kernel(
    const float* __restrict__ Wx, const float* __restrict__ Wh,
    const float* __restrict__ hidden, unsigned char* __restrict__ ws) {
  __bf16* wxt = (__bf16*)(ws + WXT_OFF);
  __bf16* wht = (__bf16*)(ws + WHT_OFF);
  __bf16* hb0 = (__bf16*)(ws + HB0_OFF);
  unsigned* sync = (unsigned*)(ws + SYNC_OFF);
  int idx = blockIdx.x * blockDim.x + threadIdx.x;
  if (idx == 0) { sync[0] = 0u; sync[1] = 0u; }     // reset grid barrier
  if (idx < H_ * H_) {                              // Wh[k][n] -> WhT[n][k]
    int k = idx / H_, n = idx % H_;
    wht[(size_t)n * H_ + k] = (__bf16)Wh[idx];
  }
  if (idx < E_ * H_) {                              // Wx[k][n] -> WxT[n][k]
    int k = idx / H_, n = idx % H_;
    wxt[(size_t)n * E_ + k] = (__bf16)Wx[idx];
  }
  if (idx < B_ * H_) hb0[idx] = (__bf16)hidden[idx];
}

// ---------------------------------------------------------------------------
// Projection: out[r][:] = emb[x[r]] @ Wx + b for all r in [0, B*S)
// Fused gather + f32->bf16 convert on the A side. Block = 8 waves covering a
// 32x256 tile; each wave = one 16-row A fragment reused across 4 B tiles.
// ---------------------------------------------------------------------------
__global__ __launch_bounds__(256) void proj_kernel(
    const int* __restrict__ x, const float* __restrict__ emb,
    const float* __restrict__ bias, const unsigned char* __restrict__ ws,
    float* __restrict__ out) {
  const __bf16* wxt = (const __bf16*)(ws + WXT_OFF);
  const int lane = threadIdx.x & 31;
  const int wave = threadIdx.x >> 5;         // 0..7
  const int mt = wave & 1;                   // 2 m-tiles (16 rows each)
  const int nq = wave >> 1;                  // 4 n-quads (64 cols each)
  const int r0 = blockIdx.x * 32 + mt * 16;
  const int c0 = blockIdx.y * 256 + nq * 64;
  const int mrow = r0 + (lane & 15);
  const int token = x[mrow];
  const float* arow = emb + (size_t)token * E_;
  const int koff = (lane >> 4) * 8;          // A lane layout: K {0..7,16..23} / {8..15,24..31}
  const int nlo = lane & 15;
  v8f acc[4] = {};

  for (int kb = 0; kb < E_; kb += 32) {
    __builtin_prefetch(arow + kb + 32, 0, 3);          // global_prefetch_b8
    V16 a;
#pragma unroll
    for (int i = 0; i < 8; ++i) {
      a.v[i]     = (__bf16)arow[kb + koff + i];
      a.v[8 + i] = (__bf16)arow[kb + koff + 16 + i];
    }
    const int kb2 = kb + (lane >> 4) * 8;    // B lane layout: K {0..7,16..23} / {8..15,24..31}
#pragma unroll
    for (int j = 0; j < 4; ++j) {
      const __bf16* bp = wxt + (size_t)(c0 + j * 16 + nlo) * E_ + kb2;
      V16 b;
      b.h[0] = *(const v8bf*)(bp);
      b.h[1] = *(const v8bf*)(bp + 16);
      acc[j] = __builtin_amdgcn_wmma_f32_16x16x32_bf16(
          false, a.v, false, b.v, (short)0, acc[j], false, false);
    }
  }
  const int m0 = (lane >> 4) * 8;            // C/D layout: m = 8*(l>>4)+v, n = l&15
#pragma unroll
  for (int j = 0; j < 4; ++j) {
    const int n = c0 + j * 16 + nlo;
    const float bv = bias[n];
#pragma unroll
    for (int v = 0; v < 8; ++v)
      out[(size_t)(r0 + m0 + v) * H_ + n] = acc[j][v] + bv;
  }
}

// ---------------------------------------------------------------------------
// Recurrent scan: 32 persistent WGs, each owns 32 columns of H with its
// WhT slice resident in LDS (64 KB). h double-buffered bf16 in L2.
// out[b][t][:] holds xw on entry to step t, h_t on exit (in-place).
// ---------------------------------------------------------------------------
__global__ __launch_bounds__(128) void scan_kernel(
    unsigned char* __restrict__ ws, float* __restrict__ out) {
  const __bf16* wht = (const __bf16*)(ws + WHT_OFF);
  __bf16* hbuf0 = (__bf16*)(ws + HB0_OFF);
  __bf16* hbuf1 = (__bf16*)(ws + HB1_OFF);
  unsigned* sync = (unsigned*)(ws + SYNC_OFF);

  __shared__ __bf16 whs[COLS_PER_WG][H_];    // 32 x 1024 bf16 = 64 KB LDS

  const int wg = blockIdx.x;
  const int c_base = wg * COLS_PER_WG;
  // Cooperative preload of this WG's WhT slice (contiguous b128 loads/stores)
  for (int i = threadIdx.x * 8; i < COLS_PER_WG * H_; i += blockDim.x * 8) {
    const int n = i / H_, k = i % H_;
    *(v8bf*)&whs[n][k] = *(const v8bf*)&wht[(size_t)(c_base + n) * H_ + k];
  }
  __syncthreads();

  const int lane = threadIdx.x & 31;
  const int wave = threadIdx.x >> 5;         // 0..3
  const int mt = wave & 1;                   // rows 0-15 / 16-31 of B
  const int nt = wave >> 1;                  // 16-col tile within slice
  const int r0 = mt * 16;
  const int c0l = nt * 16;
  const int koff = (lane >> 4) * 8;
  const int nlo = lane & 15;
  const int arow_l = r0 + nlo;
  const int bcol = c0l + nlo;
  const int m0 = (lane >> 4) * 8;
  const int n_g = c_base + c0l + nlo;

  for (int t = 0; t < S_; ++t) {
    const __bf16* cur = (t & 1) ? hbuf1 : hbuf0;
    __bf16* nxt       = (t & 1) ? hbuf0 : hbuf1;
    const __bf16* arow = cur + arow_l * H_;
    v8f acc = {};
    for (int kb = 0; kb < H_; kb += 32) {
      V16 a, b;
      a.h[0] = *(const v8bf*)(arow + kb + koff);
      a.h[1] = *(const v8bf*)(arow + kb + koff + 16);
      const __bf16* bp = &whs[bcol][kb + (lane >> 4) * 8];
      b.h[0] = *(const v8bf*)(bp);
      b.h[1] = *(const v8bf*)(bp + 16);
      acc = __builtin_amdgcn_wmma_f32_16x16x32_bf16(
          false, a.v, false, b.v, (short)0, acc, false, false);
    }
    // epilogue: h_t = tanh(xw + h@Wh); update out in-place + bf16 next-h
#pragma unroll
    for (int v = 0; v < 8; ++v) {
      const int brow = r0 + m0 + v;                        // batch index
      const size_t oidx = ((size_t)brow * S_ + t) * H_ + n_g;
      const float val = tanhf(acc[v] + out[oidx]);
      out[oidx] = val;
      nxt[brow * H_ + n_g] = (__bf16)val;
      if (t == S_ - 1)
        out[(size_t)B_ * S_ * H_ + (size_t)brow * H_ + n_g] = val;  // final state
    }
    // device-wide barrier (arrive counter + epoch flag in ws)
    __threadfence();
    __syncthreads();
    if (threadIdx.x == 0) {
      unsigned prev = __hip_atomic_fetch_add(&sync[0], 1u, __ATOMIC_ACQ_REL,
                                             __HIP_MEMORY_SCOPE_AGENT);
      if (prev == NWG - 1) {
        __hip_atomic_store(&sync[0], 0u, __ATOMIC_RELAXED,
                           __HIP_MEMORY_SCOPE_AGENT);
        __hip_atomic_store(&sync[1], (unsigned)(t + 1), __ATOMIC_RELEASE,
                           __HIP_MEMORY_SCOPE_AGENT);
      } else {
        while (__hip_atomic_load(&sync[1], __ATOMIC_ACQUIRE,
                                 __HIP_MEMORY_SCOPE_AGENT) <= (unsigned)t) {
          __builtin_amdgcn_s_sleep(1);
        }
      }
    }
    __syncthreads();
  }
}

// ---------------------------------------------------------------------------
extern "C" void kernel_launch(void* const* d_in, const int* in_sizes, int n_in,
                              void* d_out, int out_size, void* d_ws, size_t ws_size,
                              hipStream_t stream) {
  const int*   x      = (const int*)d_in[0];
  const float* hidden = (const float*)d_in[1];
  const float* emb    = (const float*)d_in[2];
  const float* Wx     = (const float*)d_in[3];
  const float* Wh     = (const float*)d_in[4];
  const float* bias   = (const float*)d_in[5];
  unsigned char* ws   = (unsigned char*)d_ws;
  float* out          = (float*)d_out;
  (void)in_sizes; (void)n_in; (void)out_size; (void)ws_size;

  // 1) weight transpose/convert + sync reset (graph-replay safe)
  prep_kernel<<<(H_ * H_ + 255) / 256, 256, 0, stream>>>(Wx, Wh, hidden, ws);

  // 2) fused embedding-gather + input projection GEMM -> out[0 : B*S*H)
  dim3 pg(B_ * S_ / 32, H_ / 256);
  proj_kernel<<<pg, 256, 0, stream>>>(x, emb, bias, ws, out);

  // 3) persistent recurrent scan (in-place over out, appends final state)
  scan_kernel<<<NWG, 128, 0, stream>>>(ws, out);
}
